// SelfAttention_2095944041081
// MI455X (gfx1250) — matrix operand order repaired and checked by reading, not compiled, not run on hardware
//
#include <hip/hip_runtime.h>
#include <hip/hip_bf16.h>

typedef __attribute__((ext_vector_type(16))) _Float16 v16h;
typedef __attribute__((ext_vector_type(8)))  float    v8f;
typedef __attribute__((ext_vector_type(4)))  float    v4f;

#define WMMA_F16(a,b,c) __builtin_amdgcn_wmma_f32_16x16x32_f16(false,(a),false,(b),(short)0,(c),false,false)

// Problem shape
constexpr int BB = 4, CC = 256, NN = 4096, DD = 32;

// Workspace layout (units: _Float16 halves). All blocks are 512-half (1KB) WMMA tiles.
constexpr size_t OFF_XH = 0;                               // x as B-tiles: [B][C/32][N/16][512]
constexpr size_t SZ_XH  = (size_t)BB * 8 * 256 * 512;      // 8 MB
constexpr size_t OFF_QW = OFF_XH + SZ_XH;                  // q_weight A-tiles: [2][8][512]
constexpr size_t OFF_KW = OFF_QW + 2 * 8 * 512;
constexpr size_t OFF_VW = OFF_KW + 2 * 8 * 512;            // v_weight A-tiles: [16][8][512]
constexpr size_t OFF_QH = OFF_VW + 16 * 8 * 512;           // q as A-tiles: [B][N/16][512] (row=i,K=d)
constexpr size_t SZ_QH  = (size_t)BB * 256 * 512;          // 1 MB
constexpr size_t OFF_KH = OFF_QH + SZ_QH;                  // k as B-tiles: [B][N/16][512] (K=d,col=j)
constexpr size_t OFF_VH = OFF_KH + SZ_QH;                  // v^T as B-tiles: [B][N/32][C/16][512] (K=j,col=c)
// total = OFF_VH + B*128*16*512 halves  ~= 18.2 MB

// ---------------------------------------------------------------------------
// Kernel 1: x (f32, (B,C,N)) -> f16 WMMA B-matrix swizzled tiles
// B-tile swizzle: lane = col + 16*((K%16)/8), e = K%8 + 8*(K/16), idx = lane*16 + e
// ---------------------------------------------------------------------------
__global__ void __launch_bounds__(256) prep_x_kernel(const float* __restrict__ x,
                                                     _Float16* __restrict__ ws) {
    int id = blockIdx.x * 256 + threadIdx.x;          // < B*C*N
    int n = id % NN;
    int c = (id / NN) % CC;
    int b = id / (NN * CC);
    float v = x[id];
    int kblk = c >> 5, K = c & 31, nt = n >> 4, col = n & 15;
    size_t idx = OFF_XH + (((size_t)(b * 8 + kblk) * 256 + nt) << 9)
               + (size_t)((col + (((K & 15) >> 3) << 4)) << 4) + ((K & 7) + ((K >> 4) << 3));
    ws[idx] = (_Float16)v;
}

// ---------------------------------------------------------------------------
// Kernel 2: weights (f32, (D,C)) -> f16 WMMA A-matrix swizzled tiles
// A-tile swizzle: lane = row + 16*((K%16)/8), e = K%8 + 8*(K/16)
// ---------------------------------------------------------------------------
__global__ void __launch_bounds__(256) prep_w_kernel(const float* __restrict__ qw,
                                                     const float* __restrict__ kw,
                                                     const float* __restrict__ vw,
                                                     _Float16* __restrict__ ws) {
    int id = blockIdx.x * 256 + threadIdx.x;          // < 8192+8192+65536
    const float* src; size_t base; int d, c;
    if (id < 8192)        { src = qw; base = OFF_QW; d = id / CC;            c = id % CC; }
    else if (id < 16384)  { src = kw; base = OFF_KW; d = (id - 8192) / CC;   c = (id - 8192) % CC; }
    else                  { src = vw; base = OFF_VW; d = (id - 16384) / CC;  c = (id - 16384) % CC; }
    float v = src[d * CC + c];
    int rt = d >> 4, row = d & 15, kblk = c >> 5, K = c & 31;
    int lane = row + (((K & 15) >> 3) << 4);
    int e    = (K & 7) + ((K >> 4) << 3);
    ws[base + (size_t)(rt * 8 + kblk) * 512 + lane * 16 + e] = (_Float16)v;
}

// ---------------------------------------------------------------------------
// Kernel 3: projections q/k/v = W @ x via WMMA. One wave per 16x16 output tile.
// Results scattered into the attention-pass swizzled layouts.
// ---------------------------------------------------------------------------
__global__ void __launch_bounds__(256) proj_kernel(_Float16* __restrict__ ws) {
    int lane = threadIdx.x & 31, wid = threadIdx.x >> 5;
    int gw = blockIdx.x * 8 + wid;                    // < 4*5120
    int b = gw / 5120, t = gw % 5120;
    const _Float16* Abase; int rt, nt, kind;
    if (t < 512)        { kind = 0; rt = t >> 8; nt = t & 255; Abase = ws + OFF_QW; }
    else if (t < 1024)  { kind = 1; t -= 512;  rt = t >> 8; nt = t & 255; Abase = ws + OFF_KW; }
    else                { kind = 2; t -= 1024; rt = t >> 8; nt = t & 255; Abase = ws + OFF_VW; }

    v8f acc = {};
    const _Float16* xb = ws + OFF_XH + ((size_t)(b * 8) * 256 + nt) * 512 + lane * 16;
#pragma unroll
    for (int kb = 0; kb < 8; ++kb) {
        v16h a  = *(const v16h*)(Abase + (size_t)(rt * 8 + kb) * 512 + lane * 16);
        v16h bm = *(const v16h*)(xb + (size_t)kb * 256 * 512);
        acc = WMMA_F16(a, bm, acc);
    }
    int hl = lane >> 4, ln = lane & 15;
#pragma unroll
    for (int r = 0; r < 8; ++r) {
        float val = acc[r];
        if (kind == 0) {            // q -> A-layout (row=i%16, K=d in [0,32))
            int d  = rt * 16 + r + 8 * hl;
            int lt = ln + (((d & 15) >> 3) << 4);
            int e  = (d & 7) + ((d >> 4) << 3);
            ws[OFF_QH + (size_t)(b * 256 + nt) * 512 + lt * 16 + e] = (_Float16)val;
        } else if (kind == 1) {     // k -> B-layout (K=d, col=j%16)
            int d = rt * 16 + r + 8 * hl;
            ws[OFF_KH + (size_t)(b * 256 + nt) * 512 + (ln + (((d & 15) >> 3) << 4)) * 16 + ((d & 7) + ((d >> 4) << 3))] = (_Float16)val;
        } else {                    // v -> B-layout (K=n%32, col=c%16)
            int cmod = r + 8 * hl;  // c_out % 16
            ws[OFF_VH + ((size_t)(b * 128 + (nt >> 1)) * 16 + rt) * 512
               + (cmod + ((ln >> 3) << 4)) * 16 + ((ln & 7) + ((nt & 1) << 3))] = (_Float16)val;
        }
    }
}

// ---------------------------------------------------------------------------
// Kernel 4: flash attention. 512 blocks (b, 32-query group) x 8 waves.
// wave = (row-tile rt in {0,1}, channel-group cg in {0..3} of 64 channels).
// Per 32-key step: 2 QK WMMAs, online softmax, C->A transpose via LDS, 4 PV WMMAs.
// ---------------------------------------------------------------------------
__global__ void __launch_bounds__(256) attn_kernel(const float* __restrict__ x,
                                                   const float* __restrict__ gamma_p,
                                                   const _Float16* __restrict__ ws,
                                                   float* __restrict__ out) {
    __shared__ __align__(32) _Float16 shP[8 * 512];
    int lane = threadIdx.x & 31, wid = threadIdx.x >> 5;
    int b  = blockIdx.x >> 7;           // batch
    int nb = blockIdx.x & 127;          // 32-query group
    int rt = wid >> 2, cg = wid & 3;
    int itile = nb * 2 + rt;            // i / 16
    int hl = lane >> 4, ln = lane & 15;

    // Q tile: A-layout, one contiguous 32B load per lane, reused all 128 steps
    v16h aq = *(const v16h*)(ws + OFF_QH + (size_t)(b * 256 + itile) * 512 + lane * 16);

    v8f acc0 = {}, acc1 = {}, acc2 = {}, acc3 = {};
    float m[8], l[8];
#pragma unroll
    for (int r = 0; r < 8; ++r) { m[r] = -1e30f; l[r] = 0.f; }

    _Float16* myP = shP + wid * 512;
    const _Float16* khb = ws + OFF_KH + (size_t)b * 256 * 512 + lane * 16;
    const _Float16* vhb = ws + OFF_VH + (size_t)b * 128 * 16 * 512 + (size_t)(cg * 4) * 512 + lane * 16;

    for (int jb = 0; jb < 128; ++jb) {
        v16h k0 = *(const v16h*)(khb + (size_t)(2 * jb) * 512);
        v16h k1 = *(const v16h*)(khb + (size_t)(2 * jb + 1) * 512);
        v8f s0 = {}, s1 = {};
        s0 = WMMA_F16(aq, k0, s0);      // S rows=i, cols=j0..15
        s1 = WMMA_F16(aq, k1, s1);      // S rows=i, cols=j16..31

        float p0[8], p1[8];
#pragma unroll
        for (int r = 0; r < 8; ++r) {
            // row-wise max over the 32-key chunk (rows live in 16-lane halves)
            float mx = fmaxf(s0[r], s1[r]);
            mx = fmaxf(mx, __shfl_xor(mx, 1));
            mx = fmaxf(mx, __shfl_xor(mx, 2));
            mx = fmaxf(mx, __shfl_xor(mx, 4));
            mx = fmaxf(mx, __shfl_xor(mx, 8));
            float mn = fmaxf(m[r], mx);
            float sc = __expf(m[r] - mn);
            m[r] = mn;
            float a0 = __expf(s0[r] - mn);
            float a1 = __expf(s1[r] - mn);
            p0[r] = a0; p1[r] = a1;
            float sum = a0 + a1;
            sum += __shfl_xor(sum, 1);
            sum += __shfl_xor(sum, 2);
            sum += __shfl_xor(sum, 4);
            sum += __shfl_xor(sum, 8);
            l[r] = l[r] * sc + sum;
            acc0[r] *= sc; acc1[r] *= sc; acc2[r] *= sc; acc3[r] *= sc;
        }

        // transpose P from C-layout to A-layout via this wave's private LDS slab
#pragma unroll
        for (int r = 0; r < 8; ++r) {
            int row = r + 8 * hl;
            int lt  = row + ((ln >> 3) << 4);   // same target lane for both halves of K
            myP[lt * 16 + (ln & 7)]     = (_Float16)p0[r];  // K = ln
            myP[lt * 16 + 8 + (ln & 7)] = (_Float16)p1[r];  // K = 16 + ln
        }
        asm volatile("s_wait_dscnt 0" ::: "memory");  // same-wave LDS store->load hazard
        v16h ap = *(const v16h*)(myP + lane * 16);

        const _Float16* vb = vhb + (size_t)jb * 16 * 512;
        v16h v0 = *(const v16h*)(vb);
        v16h v1 = *(const v16h*)(vb + 512);
        v16h v2 = *(const v16h*)(vb + 1024);
        v16h v3 = *(const v16h*)(vb + 1536);
        acc0 = WMMA_F16(ap, v0, acc0);
        acc1 = WMMA_F16(ap, v1, acc1);
        acc2 = WMMA_F16(ap, v2, acc2);
        acc3 = WMMA_F16(ap, v3, acc3);
    }

    // epilogue: out = gamma * acc / l + x, C-layout rows are contiguous in n
    float g = gamma_p[0];
    int i0 = (itile << 4) + 8 * hl;
#pragma unroll
    for (int tt = 0; tt < 4; ++tt) {
        v8f a = (tt == 0) ? acc0 : (tt == 1) ? acc1 : (tt == 2) ? acc2 : acc3;
        int c = cg * 64 + tt * 16 + ln;
        size_t base = ((size_t)(b * CC + c)) * NN + i0;
        v4f lo, hi;
#pragma unroll
        for (int r = 0; r < 4; ++r) {
            lo[r] = g * a[r]     / l[r]     + x[base + r];
            hi[r] = g * a[4 + r] / l[4 + r] + x[base + 4 + r];
        }
        *(v4f*)(out + base)     = lo;
        *(v4f*)(out + base + 4) = hi;
    }
}

// ---------------------------------------------------------------------------
extern "C" void kernel_launch(void* const* d_in, const int* in_sizes, int n_in,
                              void* d_out, int out_size, void* d_ws, size_t ws_size,
                              hipStream_t stream) {
    const float* x  = (const float*)d_in[0];
    const float* qw = (const float*)d_in[1];
    const float* kw = (const float*)d_in[2];
    const float* vw = (const float*)d_in[3];
    const float* gm = (const float*)d_in[4];
    float* out = (float*)d_out;
    _Float16* ws = (_Float16*)d_ws;

    // 1) x -> f16 B-tiles          (B*C*N = 4,194,304 elements)
    prep_x_kernel<<<16384, 256, 0, stream>>>(x, ws);
    // 2) weights -> f16 A-tiles    (81,920 elements)
    prep_w_kernel<<<320, 256, 0, stream>>>(qw, kw, vw, ws);
    // 3) q/k/v projections via WMMA (20,480 tiles, 1 wave each, 8 waves/block)
    proj_kernel<<<2560, 256, 0, stream>>>(ws);
    // 4) flash attention (B * N/32 = 512 blocks of 8 waves)
    attn_kernel<<<512, 256, 0, stream>>>(x, gm, ws, out);
}
